// FullyJITTransformerBlock_15272903704645
// MI455X (gfx1250) — compile-verified, matrix operand order checked by reading, not run on hardware
//
#include <hip/hip_runtime.h>
#include <cstdint>

typedef unsigned short u16;
typedef __attribute__((ext_vector_type(16))) __bf16 bf16x16;
typedef __attribute__((ext_vector_type(8)))  float  v8f;

#define DIM 1024
#define NHEAD 16
#define HD 64
#define MLPH 4096
#define SEQ 2048
#define BATCH 2
#define ROWS (BATCH*SEQ)          // 4096
#define SCALE 0.125f              // 64^-0.5
#define LNEPS 1e-5f

union Frag { bf16x16 v; uint4 q[2]; };

static __device__ __forceinline__ u16 f2bf(float f) {
    uint32_t u = __float_as_uint(f);
    uint32_t r = u + 0x7FFFu + ((u >> 16) & 1u);   // round-to-nearest-even
    return (u16)(r >> 16);
}

static __device__ __forceinline__ v8f wmma_bf16(bf16x16 a, bf16x16 b, v8f c) {
    return __builtin_amdgcn_wmma_f32_16x16x32_bf16(false, a, false, b, (short)0, c, false, false);
}

// ---------------------------------------------------------------- convert f32 -> bf16
__global__ __launch_bounds__(256) void cvt_bf16_kernel(const float* __restrict__ in,
                                                       u16* __restrict__ out, int n) {
    int i = blockIdx.x * 256 + threadIdx.x;
    if (i < n) out[i] = f2bf(in[i]);
}

// ---------------------------------------------------------------- LayerNorm (row of 1024) -> bf16
__global__ __launch_bounds__(256) void layernorm_bf16_kernel(const float* __restrict__ x,
                                                             const float* __restrict__ w,
                                                             const float* __restrict__ b,
                                                             u16* __restrict__ out) {
    __shared__ float r1[256], r2[256];
    const int row = blockIdx.x, t = threadIdx.x;
    const float4 v = reinterpret_cast<const float4*>(x + (size_t)row * DIM)[t];
    r1[t] = v.x + v.y + v.z + v.w;
    r2[t] = v.x*v.x + v.y*v.y + v.z*v.z + v.w*v.w;
    __syncthreads();
    for (int st = 128; st > 0; st >>= 1) {
        if (t < st) { r1[t] += r1[t + st]; r2[t] += r2[t + st]; }
        __syncthreads();
    }
    const float mean = r1[0] * (1.0f / DIM);
    const float var  = r2[0] * (1.0f / DIM) - mean * mean;
    const float rstd = rsqrtf(var + LNEPS);
    const float4 wv = reinterpret_cast<const float4*>(w)[t];
    const float4 bv = reinterpret_cast<const float4*>(b)[t];
    const float y0 = (v.x - mean) * rstd * wv.x + bv.x;
    const float y1 = (v.y - mean) * rstd * wv.y + bv.y;
    const float y2 = (v.z - mean) * rstd * wv.z + bv.z;
    const float y3 = (v.w - mean) * rstd * wv.w + bv.w;
    uint2 o;
    o.x = (uint32_t)f2bf(y0) | ((uint32_t)f2bf(y1) << 16);
    o.y = (uint32_t)f2bf(y2) | ((uint32_t)f2bf(y3) << 16);
    reinterpret_cast<uint2*>(out + (size_t)row * DIM)[t] = o;
}

// ---------------------------------------------------------------- WMMA GEMM  C[M,N] = A[M,K] @ W[N,K]^T
// Async double-buffered LDS pipeline, BK=64.
// EPI 0: store f32     EPI 1: qkv scatter (bf16 Q,K head-major; V transposed)
// EPI 2: f32 = aux+acc EPI 3: bf16 = silu(aux)*acc
#define LSTR 72   // LDS row stride in u16 (64 data + 8 pad; 144B = 9x16B)

// Issue 4 async b128 loads: 64 contiguous bytes global -> LDS (per thread).
// INST_OFFSET is added to BOTH the LDS and global addresses (GV mode), so one
// base pair covers all four chunks.
static __device__ __forceinline__ void async_tile_load(const u16* __restrict__ src,
                                                       int row0, int K, int k0,
                                                       u16* dstLds, int tid) {
    const int row = tid >> 1;
    const int colbase = (tid & 1) * 32;               // u16 units (64B half-row)
    const uint32_t l = (uint32_t)(size_t)(dstLds + row * LSTR + colbase);
    const uint64_t g = (uint64_t)(size_t)(src + (size_t)(row0 + row) * K + k0 + colbase);
    asm volatile(
        "global_load_async_to_lds_b128 %0, %1, off\n\t"
        "global_load_async_to_lds_b128 %0, %1, off offset:16\n\t"
        "global_load_async_to_lds_b128 %0, %1, off offset:32\n\t"
        "global_load_async_to_lds_b128 %0, %1, off offset:48"
        :: "v"(l), "v"(g) : "memory");
}

template <int EPI>
__global__ __launch_bounds__(256) void gemm_bf16_kernel(
    const u16* __restrict__ A, const u16* __restrict__ Bw,
    float* __restrict__ C, const float* __restrict__ aux,
    u16* __restrict__ oq, u16* __restrict__ ok, u16* __restrict__ ov,
    int M, int N, int K)
{
    __shared__ u16 As[2][128 * LSTR];
    __shared__ u16 Bs[2][128 * LSTR];
    const int tid  = threadIdx.x;
    const int lane = tid & 31, wid = tid >> 5;
    const int half = lane >> 4, lrow = lane & 15;
    const int wm = wid >> 2, wn = wid & 3;         // 2 x 4 wave grid
    const int bm = blockIdx.y * 128, bn = blockIdx.x * 128;

    v8f acc[4][2];
#pragma unroll
    for (int i = 0; i < 4; ++i)
#pragma unroll
        for (int j = 0; j < 2; ++j)
            acc[i][j] = (v8f){0.f,0.f,0.f,0.f,0.f,0.f,0.f,0.f};

    // prologue: async-load tile 0 into buffer 0
    async_tile_load(A,  bm, K, 0, &As[0][0], tid);
    async_tile_load(Bw, bn, K, 0, &Bs[0][0], tid);
    asm volatile("s_wait_asynccnt 0" ::: "memory");
    __syncthreads();

    int buf = 0;
    for (int k0 = 0; k0 < K; k0 += 64) {
        // start DMA of next tile into the other buffer (overlaps WMMA below)
        if (k0 + 64 < K) {
            async_tile_load(A,  bm, K, k0 + 64, &As[buf ^ 1][0], tid);
            async_tile_load(Bw, bn, K, k0 + 64, &Bs[buf ^ 1][0], tid);
        }
        if (k0 + 128 < K) {   // warm L2 two tiles ahead
            const int prow = tid >> 1, pcol = (tid & 1) * 32;
            __builtin_prefetch(A  + (size_t)(bm + prow) * K + k0 + 128 + pcol, 0, 1);
            __builtin_prefetch(Bw + (size_t)(bn + prow) * K + k0 + 128 + pcol, 0, 1);
        }

        const u16* Ab = &As[buf][0];
        const u16* Bb = &Bs[buf][0];
#pragma unroll
        for (int kk = 0; kk < 2; ++kk) {
            Frag af[4], bfr[2];
#pragma unroll
            for (int i = 0; i < 4; ++i) {
                const int base = (wm * 64 + i * 16 + lrow) * LSTR + kk * 32 + half * 8;
                af[i].q[0] = *reinterpret_cast<const uint4*>(&Ab[base]);
                af[i].q[1] = *reinterpret_cast<const uint4*>(&Ab[base + 16]);
            }
#pragma unroll
            for (int j = 0; j < 2; ++j) {
                const int base = (wn * 32 + j * 16 + lrow) * LSTR + kk * 32 + half * 8;
                bfr[j].q[0] = *reinterpret_cast<const uint4*>(&Bb[base]);
                bfr[j].q[1] = *reinterpret_cast<const uint4*>(&Bb[base + 16]);
            }
#pragma unroll
            for (int i = 0; i < 4; ++i)
#pragma unroll
                for (int j = 0; j < 2; ++j)
                    acc[i][j] = wmma_bf16(af[i].v, bfr[j].v, acc[i][j]);
        }
        // my async writes done; barrier makes everyone's writes visible
        asm volatile("s_wait_asynccnt 0" ::: "memory");
        __syncthreads();
        buf ^= 1;
    }

    const int rb = bm + wm * 64, cb = bn + wn * 32;
#pragma unroll
    for (int i = 0; i < 4; ++i)
#pragma unroll
        for (int j = 0; j < 2; ++j)
#pragma unroll
            for (int v = 0; v < 8; ++v) {
                const int row = rb + i * 16 + half * 8 + v;
                const int col = cb + j * 16 + lrow;
                const float val = acc[i][j][v];
                if (EPI == 0) {
                    C[(size_t)row * N + col] = val;
                } else if (EPI == 2) {
                    C[(size_t)row * N + col] = aux[(size_t)row * N + col] + val;
                } else if (EPI == 3) {
                    const float g = aux[(size_t)row * N + col];
                    const float s = g / (1.0f + __expf(-g));
                    oq[(size_t)row * N + col] = f2bf(s * val);
                } else {  // EPI == 1 : qkv scatter
                    const int b  = row >> 11, s = row & 2047;
                    const int sec = col >> 10, dc = col & 1023;
                    const int h  = dc >> 6,  d = dc & 63;
                    const size_t head = (size_t)(b * NHEAD + h);
                    if (sec == 0)
                        oq[(head * SEQ + s) * HD + d] = f2bf(val);
                    else if (sec == 1)
                        ok[(head * SEQ + s) * HD + d] = f2bf(val);
                    else
                        ov[(head * HD + d) * SEQ + s] = f2bf(val);   // V transposed [hd, S]
                }
            }
}

// ---------------------------------------------------------------- flash attention (wave = 16 q rows)
__global__ __launch_bounds__(256) void attn_kernel(const u16* __restrict__ Q,
                                                   const u16* __restrict__ Kb,
                                                   const u16* __restrict__ Vt,
                                                   u16* __restrict__ O) {
    __shared__ u16 P[8 * 16 * 64];                 // per-wave 16x64 bf16 P tile
    const int tid = threadIdx.x, lane = tid & 31, wid = tid >> 5;
    const int half = lane >> 4, lrow = lane & 15;
    const int bx = blockIdx.x;
    const int qb = bx & 15, h = (bx >> 4) & 15, b = bx >> 8;
    const size_t headoff = (size_t)(b * NHEAD + h) * SEQ * HD;
    const u16* Qh  = Q  + headoff;
    const u16* Kh  = Kb + headoff;
    const u16* Vth = Vt + headoff;                 // [hd=64, SEQ] per head
    const int q0 = qb * 128 + wid * 16;
    u16* Pw = P + wid * 1024;

    Frag aQ[2];
#pragma unroll
    for (int ks = 0; ks < 2; ++ks) {
        const size_t ad = (size_t)(q0 + lrow) * HD + ks * 32 + half * 8;
        aQ[ks].q[0] = *reinterpret_cast<const uint4*>(Qh + ad);
        aQ[ks].q[1] = *reinterpret_cast<const uint4*>(Qh + ad + 16);
    }

    v8f o[4];
    float mrow[8], lrw[8], alpha[8];
#pragma unroll
    for (int nt = 0; nt < 4; ++nt) o[nt] = (v8f){0.f,0.f,0.f,0.f,0.f,0.f,0.f,0.f};
#pragma unroll
    for (int v = 0; v < 8; ++v) { mrow[v] = -1e30f; lrw[v] = 0.f; }

    for (int kb = 0; kb < SEQ; kb += 64) {
        v8f sc[4];
#pragma unroll
        for (int kt = 0; kt < 4; ++kt) {
            v8f c = (v8f){0.f,0.f,0.f,0.f,0.f,0.f,0.f,0.f};
#pragma unroll
            for (int ks = 0; ks < 2; ++ks) {
                Frag bK;
                const size_t ad = (size_t)(kb + kt * 16 + lrow) * HD + ks * 32 + half * 8;
                bK.q[0] = *reinterpret_cast<const uint4*>(Kh + ad);
                bK.q[1] = *reinterpret_cast<const uint4*>(Kh + ad + 16);
                c = wmma_bf16(aQ[ks].v, bK.v, c);
            }
            sc[kt] = c;
        }
        // online softmax over the wave's 16 rows (rows split: lanes<16 -> v, lanes>=16 -> v+8)
#pragma unroll
        for (int v = 0; v < 8; ++v) {
#pragma unroll
            for (int kt = 0; kt < 4; ++kt) sc[kt][v] *= SCALE;
            float mx = fmaxf(fmaxf(sc[0][v], sc[1][v]), fmaxf(sc[2][v], sc[3][v]));
#pragma unroll
            for (int off = 8; off > 0; off >>= 1) mx = fmaxf(mx, __shfl_xor(mx, off, 32));
            const float mn = fmaxf(mrow[v], mx);
            const float al = __expf(mrow[v] - mn);
            mrow[v] = mn; alpha[v] = al;
            float rs = 0.f;
#pragma unroll
            for (int kt = 0; kt < 4; ++kt) {
                const float p = __expf(sc[kt][v] - mn);
                sc[kt][v] = p; rs += p;
            }
#pragma unroll
            for (int off = 8; off > 0; off >>= 1) rs += __shfl_xor(rs, off, 32);
            lrw[v] = lrw[v] * al + rs;
        }
#pragma unroll
        for (int nt = 0; nt < 4; ++nt)
#pragma unroll
            for (int v = 0; v < 8; ++v) o[nt][v] *= alpha[v];
        // P (C-layout) -> LDS -> A-layout bf16 fragments
#pragma unroll
        for (int kt = 0; kt < 4; ++kt)
#pragma unroll
            for (int v = 0; v < 8; ++v)
                Pw[(half * 8 + v) * 64 + kt * 16 + lrow] = f2bf(sc[kt][v]);
        __syncthreads();
        Frag aP[2];
#pragma unroll
        for (int ks = 0; ks < 2; ++ks) {
            const int ad = lrow * 64 + ks * 32 + half * 8;
            aP[ks].q[0] = *reinterpret_cast<const uint4*>(&Pw[ad]);
            aP[ks].q[1] = *reinterpret_cast<const uint4*>(&Pw[ad + 16]);
        }
#pragma unroll
        for (int ot = 0; ot < 4; ++ot)
#pragma unroll
            for (int ks = 0; ks < 2; ++ks) {
                Frag bV;
                const size_t ad = (size_t)(ot * 16 + lrow) * SEQ + kb + ks * 32 + half * 8;
                bV.q[0] = *reinterpret_cast<const uint4*>(Vth + ad);
                bV.q[1] = *reinterpret_cast<const uint4*>(Vth + ad + 16);
                o[ot] = wmma_bf16(aP[ks].v, bV.v, o[ot]);
            }
        __syncthreads();
    }
    // write attn output as bf16 rows [B,S,D]
#pragma unroll
    for (int ot = 0; ot < 4; ++ot)
#pragma unroll
        for (int v = 0; v < 8; ++v) {
            const int s = q0 + half * 8 + v;
            const int d = h * HD + ot * 16 + lrow;
            O[(size_t)(b * SEQ + s) * DIM + d] = f2bf(o[ot][v] / lrw[v]);
        }
}

// ---------------------------------------------------------------- host launcher
extern "C" void kernel_launch(void* const* d_in, const int* in_sizes, int n_in,
                              void* d_out, int out_size, void* d_ws, size_t ws_size,
                              hipStream_t stream) {
    const float* x    = (const float*)d_in[0];
    const float* wqkv = (const float*)d_in[1];
    const float* wout = (const float*)d_in[2];
    const float* n1w  = (const float*)d_in[3];
    const float* n1b  = (const float*)d_in[4];
    const float* wg   = (const float*)d_in[5];
    const float* wu   = (const float*)d_in[6];
    const float* wd   = (const float*)d_in[7];
    const float* n2w  = (const float*)d_in[8];
    const float* n2b  = (const float*)d_in[9];
    float* out = (float*)d_out;

    char* ws = (char*)d_ws;
    size_t off = 0;
    auto carve = [&](size_t bytes) { char* p = ws + off; off += bytes; return p; };
    u16*   wq_bf   = (u16*)  carve((size_t)3 * DIM * DIM * 2);       // 6 MB
    u16*   wo_bf   = (u16*)  carve((size_t)DIM * DIM * 2);           // 2 MB
    u16*   wg_bf   = (u16*)  carve((size_t)MLPH * DIM * 2);          // 8 MB
    u16*   wu_bf   = (u16*)  carve((size_t)MLPH * DIM * 2);
    u16*   wd_bf   = (u16*)  carve((size_t)DIM * MLPH * 2);
    u16*   xn_bf   = (u16*)  carve((size_t)ROWS * DIM * 2);
    u16*   Qb      = (u16*)  carve((size_t)ROWS * DIM * 2);
    u16*   Kb      = (u16*)  carve((size_t)ROWS * DIM * 2);
    u16*   Vtb     = (u16*)  carve((size_t)ROWS * DIM * 2);
    u16*   aO_bf   = (u16*)  carve((size_t)ROWS * DIM * 2);
    float* x2      = (float*)carve((size_t)ROWS * DIM * 4);          // 16 MB
    u16*   xn2_bf  = (u16*)  carve((size_t)ROWS * DIM * 2);
    float* gate    = (float*)carve((size_t)ROWS * MLPH * 4);         // 64 MB
    u16*   hid_bf  = (u16*)  carve((size_t)ROWS * MLPH * 2);         // 32 MB

    const dim3 blk(256);
    auto cvt = [&](const float* src, u16* dst, int n) {
        cvt_bf16_kernel<<<dim3((n + 255) / 256), blk, 0, stream>>>(src, dst, n);
    };
    cvt(wqkv, wq_bf, 3 * DIM * DIM);
    cvt(wout, wo_bf, DIM * DIM);
    cvt(wg,   wg_bf, MLPH * DIM);
    cvt(wu,   wu_bf, MLPH * DIM);
    cvt(wd,   wd_bf, DIM * MLPH);

    // LN1
    layernorm_bf16_kernel<<<dim3(ROWS), blk, 0, stream>>>(x, n1w, n1b, xn_bf);
    // QKV GEMM with head-major scatter (V transposed)
    gemm_bf16_kernel<1><<<dim3(3 * DIM / 128, ROWS / 128), blk, 0, stream>>>(
        xn_bf, wq_bf, nullptr, nullptr, Qb, Kb, Vtb, ROWS, 3 * DIM, DIM);
    // attention
    attn_kernel<<<dim3(BATCH * NHEAD * (SEQ / 128)), blk, 0, stream>>>(Qb, Kb, Vtb, aO_bf);
    // out projection + residual
    gemm_bf16_kernel<2><<<dim3(DIM / 128, ROWS / 128), blk, 0, stream>>>(
        aO_bf, wo_bf, x2, x, nullptr, nullptr, nullptr, ROWS, DIM, DIM);
    // LN2
    layernorm_bf16_kernel<<<dim3(ROWS), blk, 0, stream>>>(x2, n2w, n2b, xn2_bf);
    // gate GEMM
    gemm_bf16_kernel<0><<<dim3(MLPH / 128, ROWS / 128), blk, 0, stream>>>(
        xn2_bf, wg_bf, gate, nullptr, nullptr, nullptr, nullptr, ROWS, MLPH, DIM);
    // up GEMM fused with silu(gate)*up -> bf16 hidden
    gemm_bf16_kernel<3><<<dim3(MLPH / 128, ROWS / 128), blk, 0, stream>>>(
        xn2_bf, wu_bf, nullptr, gate, hid_bf, nullptr, nullptr, ROWS, MLPH, DIM);
    // down GEMM + residual -> output
    gemm_bf16_kernel<2><<<dim3(DIM / 128, ROWS / 128), blk, 0, stream>>>(
        hid_bf, wd_bf, out, x2, nullptr, nullptr, nullptr, ROWS, DIM, MLPH);
    (void)in_sizes; (void)n_in; (void)out_size; (void)ws_size;
}
